// GNNModel_71459665871548
// MI455X (gfx1250) — matrix operand order repaired and checked by reading, not compiled, not run on hardware
//
#include <hip/hip_runtime.h>

// ---------------------------------------------------------------------------
// GNN forward for MI455X (gfx1250, wave32, WMMA + async-to-LDS staging).
//
//  * All GEMMs: v_wmma_f32_16x16x32_f16 (f16 A/B, f32 accumulate). h in f16
//    (16 MB) is L2-resident (192 MB L2) so per-edge gathers hit L2.
//  * Fused edge pipeline (gather -> msg1 -> lrelu -> msg2 -> lrelu ->
//    scatter-add) in one kernel: avoids ~1 GB/conv HBM round trip for the
//    [E,5,32] message tensor. GEMM1->GEMM2 fragment transpose via LDS.
//  * Weight B-fragments are pre-packed into WMMA lane layout, then staged
//    per-workgroup into LDS with GLOBAL_LOAD_ASYNC_TO_LDS_B128 (ASYNCcnt +
//    s_wait_asynccnt) so the WMMA K-loop reads B from LDS, not global.
//  * BatchNorm: two-pass f32 (LDS-reduced atomics + elementwise apply).
//
// Input order = JAX pytree flatten (dicts sorted by key): see round-0 map.
// ---------------------------------------------------------------------------

typedef __attribute__((ext_vector_type(16))) _Float16 v16h;
typedef __attribute__((ext_vector_type(8)))  _Float16 v8h;
typedef __attribute__((ext_vector_type(8)))  float    v8f;
typedef __attribute__((ext_vector_type(4)))  float    v4f;
typedef __attribute__((ext_vector_type(4)))  int      v4i;

// AS-qualified pointee typedefs for the async-to-LDS builtin
// (clang prints AS1 as __device__, AS3 as __shared__ in HIP diagnostics)
typedef __attribute__((address_space(1))) v4i gv4i;
typedef __attribute__((address_space(3))) v4i lv4i;

union UA { v16h v; v8h h[2]; _Float16 e[16]; };
union UC { v8f  v; float e[8]; };

#define BN_EPS 1e-5f

#if __has_builtin(__builtin_amdgcn_global_load_async_to_lds_b128)
#define HAVE_ASYNC_LDS 1
#endif

__device__ __forceinline__ v8f wmma16x16x32(v16h a, v16h b, v8f c) {
  return __builtin_amdgcn_wmma_f32_16x16x32_f16(false, a, false, b,
                                                (short)0, c, false, false);
}
__device__ __forceinline__ float lrelu(float x) { return x > 0.f ? x : 0.01f * x; }

// 16-byte global -> LDS copy chunk (async on gfx1250 toolchains that have it)
__device__ __forceinline__ void copy16_g2l(const void* g, void* l) {
#ifdef HAVE_ASYNC_LDS
  __builtin_amdgcn_global_load_async_to_lds_b128((gv4i*)g, (lv4i*)l, 0, 0);
#else
  *(v8h*)l = *(const v8h*)g;
#endif
}
// complete this block's staging (async join + workgroup barrier)
__device__ __forceinline__ void stage_join() {
#ifdef HAVE_ASYNC_LDS
#if __has_builtin(__builtin_amdgcn_s_wait_asynccnt)
  __builtin_amdgcn_s_wait_asynccnt(0);
#else
  asm volatile("s_wait_asynccnt 0x0" ::: "memory");
#endif
#endif
  __syncthreads();
}

// --------------------------- weight packing --------------------------------
// dst tile t = kt*(Nout/16) + nt ; 32 lanes x 16 halves each (1 KB / tile).
__global__ void pack_w_kernel(const float* __restrict__ W, int Kreal, int Nout,
                              int Kpad, _Float16* __restrict__ dst) {
  int idx = blockIdx.x * blockDim.x + threadIdx.x;
  int ntiles = Nout >> 4;
  int total = (Kpad >> 5) * ntiles * 32;
  if (idx >= total) return;
  int lane = idx & 31, t = idx >> 5;
  int kt = t / ntiles, nt = t - kt * ntiles;
  int n = nt * 16 + (lane & 15), kh = lane >> 4;
  _Float16* o = dst + (size_t)t * 512 + lane * 16;
  for (int j = 0; j < 16; ++j) {
    int k = kt * 32 + (j < 8 ? kh * 8 + j : 16 + kh * 8 + (j - 8));
    float w = (k < Kreal) ? W[(size_t)k * Nout + n] : 0.f;
    o[j] = (_Float16)w;
  }
}

// --------------------------- embeddings ------------------------------------
__global__ void embed_nodes_kernel(const float* __restrict__ nf,
                                   const float* __restrict__ W,  // [7,32]
                                   const float* __restrict__ b,
                                   float* __restrict__ h32,
                                   _Float16* __restrict__ h16, long n) {
  long i = (long)blockIdx.x * blockDim.x + threadIdx.x;
  if (i >= n) return;
  long r = i >> 5; int o = (int)(i & 31);
  const float* x = nf + r * 7;
  float a = b[o];
  for (int k = 0; k < 7; ++k) a += x[k] * W[k * 32 + o];
  h32[i] = a;
  h16[i] = (_Float16)a;
}

__global__ void embed_edges_kernel(const float* __restrict__ ef,
                                   const float* __restrict__ W,  // [2,8]
                                   const float* __restrict__ b,
                                   _Float16* __restrict__ ea, long n) {
  long i = (long)blockIdx.x * blockDim.x + threadIdx.x;
  if (i >= n) return;
  long r = i >> 3; int o = (int)(i & 7);
  float a = b[o] + ef[r * 2] * W[o] + ef[r * 2 + 1] * W[8 + o];
  ea[i] = (_Float16)a;
}

__global__ void zero_kernel(float* __restrict__ p, long n) {
  long i = (long)blockIdx.x * blockDim.x + threadIdx.x;
  if (i < n) p[i] = 0.f;
}

// ------------------- fused edge conv (msg1+msg2+scatter) -------------------
// One wave: 16 edge-rows x 32 outputs, 8 WMMAs. Weights staged in LDS.
__global__ void __launch_bounds__(256) edge_conv_kernel(
    const _Float16* __restrict__ h16, const _Float16* __restrict__ ea16,
    const int* __restrict__ src, const int* __restrict__ dst,
    const _Float16* __restrict__ w1, const float* __restrict__ b1,
    const _Float16* __restrict__ w2, const float* __restrict__ b2,
    float* __restrict__ agg) {
  __shared__ _Float16 wsh[4096] __attribute__((aligned(32)));          // 8 KB
  __shared__ _Float16 stage[8][16 * 40] __attribute__((aligned(16)));  // 10 KB

  // stage msg1 (6 KB) + msg2 (2 KB) B-tiles into LDS, async
  for (int ci = threadIdx.x; ci < 512; ci += 256) {
    const char* srcp = (ci < 384) ? (const char*)w1 + ci * 16
                                  : (const char*)w2 + (ci - 384) * 16;
    copy16_g2l(srcp, (char*)wsh + ci * 16);
  }
  stage_join();

  const int lane = threadIdx.x & 31;
  const int wid  = threadIdx.x >> 5;
  const int m  = lane & 15;
  const int kh = lane >> 4;
  const int base = (blockIdx.x * 8 + wid) * 16;  // rows_e = 4,000,000 exact

  // gather A fragments: this lane owns tile row m
  int ia = base + m;
  int e = ia / 5, c = ia - e * 5;
  int rd = dst[e] * 5 + c;
  int rs = src[e] * 5 + c;
  UA aD, aS, aE;
  aD.h[0] = *(const v8h*)(h16 + (size_t)rd * 32 + kh * 8);
  aD.h[1] = *(const v8h*)(h16 + (size_t)rd * 32 + 16 + kh * 8);
  aS.h[0] = *(const v8h*)(h16 + (size_t)rs * 32 + kh * 8);
  aS.h[1] = *(const v8h*)(h16 + (size_t)rs * 32 + 16 + kh * 8);
  for (int j = 0; j < 16; ++j) aE.e[j] = (_Float16)0.f;
  if (kh == 0) aE.h[0] = *(const v8h*)(ea16 + (size_t)ia * 8);  // K 0..7 real

  // GEMM1: [16 x 96(pad 72)] @ [96 x 32]
  UC acc[2];
  for (int nt = 0; nt < 2; ++nt) {
    float bv = b1[nt * 16 + m];
    for (int j = 0; j < 8; ++j) acc[nt].e[j] = bv;
  }
  for (int nt = 0; nt < 2; ++nt) {
    v16h B0 = *(const v16h*)(wsh + (size_t)(0 * 2 + nt) * 512 + lane * 16);
    v16h B1 = *(const v16h*)(wsh + (size_t)(1 * 2 + nt) * 512 + lane * 16);
    v16h B2 = *(const v16h*)(wsh + (size_t)(2 * 2 + nt) * 512 + lane * 16);
    acc[nt].v = wmma16x16x32(aD.v, B0, acc[nt].v);
    acc[nt].v = wmma16x16x32(aS.v, B1, acc[nt].v);
    acc[nt].v = wmma16x16x32(aE.v, B2, acc[nt].v);
  }

  // lrelu, then C-layout -> A-layout transpose via LDS (in-order per wave)
  _Float16* st = stage[wid];
  for (int nt = 0; nt < 2; ++nt)
    for (int v = 0; v < 8; ++v) {
      float x = lrelu(acc[nt].e[v]);
      st[(kh * 8 + v) * 40 + nt * 16 + m] = (_Float16)x;
    }
  asm volatile("" ::: "memory");
  UA a2;
  a2.h[0] = *(const v8h*)(st + m * 40 + kh * 8);
  a2.h[1] = *(const v8h*)(st + m * 40 + 16 + kh * 8);

  // GEMM2: [16 x 32] @ [32 x 32]  (w2 tiles live at wsh + 3072 halves)
  UC acc2[2];
  for (int nt = 0; nt < 2; ++nt) {
    float bv = b2[nt * 16 + m];
    for (int j = 0; j < 8; ++j) acc2[nt].e[j] = bv;
    v16h B = *(const v16h*)(wsh + 3072 + (size_t)nt * 512 + lane * 16);
    acc2[nt].v = wmma16x16x32(a2.v, B, acc2[nt].v);
  }

  // lrelu + segment-sum scatter (f32 atomics)
  int rowi[8];
  for (int v = 0; v < 8; ++v) {
    int i2 = base + kh * 8 + v;
    int e2 = i2 / 5, c2 = i2 - e2 * 5;
    rowi[v] = dst[e2] * 5 + c2;
  }
  for (int nt = 0; nt < 2; ++nt) {
    int col = nt * 16 + m;
    for (int v = 0; v < 8; ++v)
      atomicAdd(&agg[(size_t)rowi[v] * 32 + col], lrelu(acc2[nt].e[v]));
  }
}

// ------------------------ generic f16 WMMA GEMM ----------------------------
// Block = 8 waves sharing one 32-col output pair (np); B tiles staged in LDS.
// grid.x = (Nout/32) * rowBlocks, rowBlocks = ceil(rows/16/8).
__global__ void __launch_bounds__(256) gemm_f16_kernel(
    const _Float16* __restrict__ A, const _Float16* __restrict__ wt,
    const float* __restrict__ bias, float* __restrict__ C,
    int rows, int K, int Nout) {
  __shared__ _Float16 wlds[8192] __attribute__((aligned(32)));  // up to 16 KB
  const int rowTiles  = rows >> 4;
  const int rowBlocks = (rowTiles + 7) >> 3;
  const int np = blockIdx.x / rowBlocks;
  const int rb = blockIdx.x - np * rowBlocks;
  const int ntiles = Nout >> 4;
  const int ktCount = K >> 5;

  // stage this block's B tiles: ktCount x 2 tiles of 1 KB
  for (int idx = threadIdx.x; idx < ktCount * 128; idx += 256) {
    int kt = idx >> 7;
    int off = (idx & 127) * 16;
    const char* s = (const char*)(wt + ((size_t)kt * ntiles + np * 2) * 512) + off;
    copy16_g2l(s, (char*)wlds + kt * 2048 + off);
  }
  stage_join();

  const int lane = threadIdx.x & 31;
  const int wid  = threadIdx.x >> 5;
  const int rt = rb * 8 + wid;
  if (rt >= rowTiles) return;  // wave-uniform; barrier already passed
  const int m = lane & 15, kh = lane >> 4;
  const _Float16* Ar = A + (size_t)(rt * 16 + m) * K;

  UC acc[2];
  for (int nt = 0; nt < 2; ++nt) {
    float bv = bias[np * 32 + nt * 16 + m];
    for (int j = 0; j < 8; ++j) acc[nt].e[j] = bv;
  }
  for (int kt = 0; kt < ktCount; ++kt) {
    UA a;
    a.h[0] = *(const v8h*)(Ar + kt * 32 + kh * 8);
    a.h[1] = *(const v8h*)(Ar + kt * 32 + 16 + kh * 8);
    for (int nt = 0; nt < 2; ++nt) {
      v16h B = *(const v16h*)(wlds + ((size_t)kt * 2 + nt) * 512 + lane * 16);
      acc[nt].v = wmma16x16x32(a.v, B, acc[nt].v);
    }
  }
  const int rb16 = rt * 16 + kh * 8;
  for (int nt = 0; nt < 2; ++nt) {
    const int col = np * 32 + nt * 16 + m;
    for (int v = 0; v < 8; ++v)
      C[(size_t)(rb16 + v) * Nout + col] = acc[nt].e[v];
  }
}

// ---------------- upd1 GEMM: A = concat(h16, agg_f32) K=64 -----------------
__global__ void __launch_bounds__(256) gemm_upd1_kernel(
    const _Float16* __restrict__ h16, const float* __restrict__ agg,
    const _Float16* __restrict__ wt, const float* __restrict__ bias,
    float* __restrict__ C, int rows) {
  __shared__ _Float16 wlds[2048] __attribute__((aligned(32)));  // 4 KB
  for (int idx = threadIdx.x; idx < 256; idx += 256)
    copy16_g2l((const char*)wt + idx * 16, (char*)wlds + idx * 16);
  stage_join();

  const int lane = threadIdx.x & 31, wid = threadIdx.x >> 5;
  const int tiles = rows >> 4;
  const int gw = blockIdx.x * 8 + wid;
  if (gw >= tiles) return;
  const int m = lane & 15, kh = lane >> 4;
  const int row = gw * 16 + m;

  UA a0, a1;
  a0.h[0] = *(const v8h*)(h16 + (size_t)row * 32 + kh * 8);
  a0.h[1] = *(const v8h*)(h16 + (size_t)row * 32 + 16 + kh * 8);
  const float* ag = agg + (size_t)row * 32;
  v4f f0 = *(const v4f*)(ag + kh * 8);
  v4f f1 = *(const v4f*)(ag + kh * 8 + 4);
  v4f f2 = *(const v4f*)(ag + 16 + kh * 8);
  v4f f3 = *(const v4f*)(ag + 16 + kh * 8 + 4);
  for (int j = 0; j < 4; ++j) {
    a1.e[j]      = (_Float16)f0[j];
    a1.e[4 + j]  = (_Float16)f1[j];
    a1.e[8 + j]  = (_Float16)f2[j];
    a1.e[12 + j] = (_Float16)f3[j];
  }
  UC acc[2];
  for (int nt = 0; nt < 2; ++nt) {
    float bv = bias[nt * 16 + m];
    for (int j = 0; j < 8; ++j) acc[nt].e[j] = bv;
    v16h B0 = *(const v16h*)(wlds + (size_t)(0 * 2 + nt) * 512 + lane * 16);
    v16h B1 = *(const v16h*)(wlds + (size_t)(1 * 2 + nt) * 512 + lane * 16);
    acc[nt].v = wmma16x16x32(a0.v, B0, acc[nt].v);
    acc[nt].v = wmma16x16x32(a1.v, B1, acc[nt].v);
  }
  const int rb = gw * 16 + kh * 8;
  for (int nt = 0; nt < 2; ++nt) {
    int col = nt * 16 + m;
    for (int v = 0; v < 8; ++v)
      C[(size_t)(rb + v) * 32 + col] = acc[nt].e[v];
  }
}

// ---------------------------- BN statistics --------------------------------
__global__ void stats_c_kernel(const float* __restrict__ z, long n,
                               float* __restrict__ st) {
  __shared__ float s[10];
  if (threadIdx.x < 10) s[threadIdx.x] = 0.f;
  __syncthreads();
  long stride = (long)gridDim.x * blockDim.x;
  for (long i = (long)blockIdx.x * blockDim.x + threadIdx.x; i < n; i += stride) {
    float v = z[i];
    int ch = (int)((i >> 5) % 5);
    atomicAdd(&s[ch], v);
    atomicAdd(&s[5 + ch], v * v);
  }
  __syncthreads();
  if (threadIdx.x < 5) {
    atomicAdd(&st[threadIdx.x], s[threadIdx.x]);
    atomicAdd(&st[512 + threadIdx.x], s[5 + threadIdx.x]);
  }
}

__global__ void stats_f_kernel(const float* __restrict__ z, long n, int cols,
                               float* __restrict__ st) {
  __shared__ float s[512];
  for (int j = threadIdx.x; j < 2 * cols; j += blockDim.x) s[j] = 0.f;
  __syncthreads();
  long stride = (long)gridDim.x * blockDim.x;
  for (long i = (long)blockIdx.x * blockDim.x + threadIdx.x; i < n; i += stride) {
    float v = z[i];
    int ch = (int)(i & (cols - 1));
    atomicAdd(&s[ch], v);
    atomicAdd(&s[cols + ch], v * v);
  }
  __syncthreads();
  for (int j = threadIdx.x; j < cols; j += blockDim.x) {
    atomicAdd(&st[j], s[j]);
    atomicAdd(&st[512 + j], s[cols + j]);
  }
}

// ---------------------------- BN apply -------------------------------------
__global__ void bn_c_act_kernel(const float* __restrict__ z,
                                const float* __restrict__ st,
                                const float* __restrict__ g,
                                const float* __restrict__ b,
                                _Float16* __restrict__ z16, long n) {
  long i = (long)blockIdx.x * blockDim.x + threadIdx.x;
  if (i >= n) return;
  const float cnt = (float)(n / 5);
  int ch = (int)((i >> 5) % 5);
  float mu = st[ch] / cnt;
  float var = st[512 + ch] / cnt - mu * mu;
  float x = (z[i] - mu) * rsqrtf(var + BN_EPS) * g[ch] + b[ch];
  z16[i] = (_Float16)lrelu(x);
}

__global__ void bn_c_resid_kernel(const float* __restrict__ z,
                                  const float* __restrict__ st,
                                  const float* __restrict__ g,
                                  const float* __restrict__ b,
                                  float* __restrict__ h32,
                                  _Float16* __restrict__ h16, long n) {
  long i = (long)blockIdx.x * blockDim.x + threadIdx.x;
  if (i >= n) return;
  const float cnt = (float)(n / 5);
  int ch = (int)((i >> 5) % 5);
  float mu = st[ch] / cnt;
  float var = st[512 + ch] / cnt - mu * mu;
  float u = lrelu((z[i] - mu) * rsqrtf(var + BN_EPS) * g[ch] + b[ch]);
  float nh = h32[i] + u;  // residual
  h32[i] = nh;
  h16[i] = (_Float16)nh;
}

__global__ void bn_f_act_kernel(float* __restrict__ z,
                                const float* __restrict__ st,
                                const float* __restrict__ g,
                                const float* __restrict__ b,
                                _Float16* __restrict__ z16, long n, int cols,
                                float cnt) {
  long i = (long)blockIdx.x * blockDim.x + threadIdx.x;
  if (i >= n) return;
  int ch = (int)(i & (cols - 1));
  float mu = st[ch] / cnt;
  float var = st[512 + ch] / cnt - mu * mu;
  float x = lrelu((z[i] - mu) * rsqrtf(var + BN_EPS) * g[ch] + b[ch]);
  z[i] = x;
  if (z16) z16[i] = (_Float16)x;
}

__global__ void mean64_kernel(const float* __restrict__ z,
                              float* __restrict__ out, int rows) {
  int i = blockIdx.x * blockDim.x + threadIdx.x;
  if (i >= rows) return;
  float s = 0.f;
  for (int j = 0; j < 64; ++j) s += z[(size_t)i * 64 + j];
  out[i] = s * (1.f / 64.f);
}

// ---------------------------------------------------------------------------
extern "C" void kernel_launch(void* const* d_in, const int* in_sizes, int n_in,
                              void* d_out, int out_size, void* d_ws,
                              size_t ws_size, hipStream_t stream) {
  (void)in_sizes; (void)n_in; (void)out_size; (void)ws_size;
  const long ROWS_N = 250000;   // 50000 * 5
  const long ROWS_E = 4000000;  // 800000 * 5

  const float* node_feat = (const float*)d_in[0];
  const float* edge_feat = (const float*)d_in[1];
  const int*   ei        = (const int*)d_in[2];
  const int*   srcp = ei;            // edge_index[0]
  const int*   dstp = ei + 800000;   // edge_index[1]

  const float* edgeW = (const float*)d_in[39];
  const float* edgeB = (const float*)d_in[40];
  const float* mlp0BnB = (const float*)d_in[41];
  const float* mlp0BnG = (const float*)d_in[42];
  const float* mlp0W   = (const float*)d_in[43];
  const float* mlp0B   = (const float*)d_in[44];
  const float* mlp1BnB = (const float*)d_in[45];
  const float* mlp1BnG = (const float*)d_in[46];
  const float* mlp1W   = (const float*)d_in[47];
  const float* mlp1B   = (const float*)d_in[48];
  const float* nodeW   = (const float*)d_in[49];
  const float* nodeB   = (const float*)d_in[50];
  const float* poolBnB = (const float*)d_in[51];
  const float* poolBnG = (const float*)d_in[52];
  const float* poolW   = (const float*)d_in[53];
  const float* poolB   = (const float*)d_in[54];

  // workspace layout (bytes)
  char* ws = (char*)d_ws;
  _Float16* Wh   = (_Float16*)(ws + 0);                    // packed weights
  float*    stats= (float*)(ws + ((size_t)2  << 20));
  float*    h32  = (float*)(ws + ((size_t)4  << 20));      // 32 MB
  _Float16* h16  = (_Float16*)(ws + ((size_t)36 << 20));   // 16 MB
  _Float16* ea16 = (_Float16*)(ws + ((size_t)52 << 20));   // 64 MB
  float*    agg  = (float*)(ws + ((size_t)116 << 20));     // 32 MB
  float*    zA   = (float*)(ws + ((size_t)148 << 20));     // 52 MB
  _Float16* zA16 = (_Float16*)(ws + ((size_t)200 << 20));  // 26 MB
  float*    zB   = (float*)(ws + ((size_t)226 << 20));     // 32 MB
  _Float16* zB16 = (_Float16*)(ws + ((size_t)258 << 20));  // 13 MB

  auto cdiv = [](long a, int b) { return (int)((a + b - 1) / b); };
  auto gemmGrid = [](int rows, int Nout) {
    int rowBlocks = ((rows >> 4) + 7) >> 3;
    return (Nout >> 5) * rowBlocks;
  };
  auto pack = [&](const float* W, int Kr, int Kp, int N, size_t off_halves) {
    int total = (Kp / 32) * (N / 16) * 32;
    pack_w_kernel<<<cdiv(total, 256), 256, 0, stream>>>(W, Kr, N, Kp,
                                                        Wh + off_halves);
  };

  // packed-weight offsets in halves: per conv = 7168
  for (int i = 0; i < 3; ++i) {
    const int cb = 3 + 12 * i;
    pack((const float*)d_in[cb + 4], 72, 96, 32, (size_t)i * 7168 + 0);
    pack((const float*)d_in[cb + 6], 32, 32, 32, (size_t)i * 7168 + 3072);
    pack((const float*)d_in[cb + 8], 64, 64, 32, (size_t)i * 7168 + 4096);
    pack((const float*)d_in[cb + 10], 32, 32, 32, (size_t)i * 7168 + 6144);
  }
  pack(poolW, 160, 160, 256, 21504);
  pack(mlp0W, 256, 256, 128, 62464);
  pack(mlp1W, 128, 128, 64, 95232);

  // embeddings
  embed_nodes_kernel<<<cdiv(ROWS_N * 32, 256), 256, 0, stream>>>(
      node_feat, nodeW, nodeB, h32, h16, ROWS_N * 32);
  embed_edges_kernel<<<cdiv(ROWS_E * 8, 256), 256, 0, stream>>>(
      edge_feat, edgeW, edgeB, ea16, ROWS_E * 8);

  // conv layers
  for (int i = 0; i < 3; ++i) {
    const int cb = 3 + 12 * i;
    const float* bn1b = (const float*)d_in[cb + 0];
    const float* bn1g = (const float*)d_in[cb + 1];
    const float* bn2b = (const float*)d_in[cb + 2];
    const float* bn2g = (const float*)d_in[cb + 3];
    const float* msg1b = (const float*)d_in[cb + 5];
    const float* msg2b = (const float*)d_in[cb + 7];
    const float* upd1b = (const float*)d_in[cb + 9];
    const float* upd2b = (const float*)d_in[cb + 11];
    const _Float16* w1  = Wh + (size_t)i * 7168;
    const _Float16* w2  = Wh + (size_t)i * 7168 + 3072;
    const _Float16* wu1 = Wh + (size_t)i * 7168 + 4096;
    const _Float16* wu2 = Wh + (size_t)i * 7168 + 6144;

    zero_kernel<<<cdiv(ROWS_N * 32, 256), 256, 0, stream>>>(agg, ROWS_N * 32);
    edge_conv_kernel<<<(int)(ROWS_E / 128), 256, 0, stream>>>(
        h16, ea16, srcp, dstp, w1, msg1b, w2, msg2b, agg);

    gemm_upd1_kernel<<<cdiv(ROWS_N / 16, 8), 256, 0, stream>>>(
        h16, agg, wu1, upd1b, zA, (int)ROWS_N);
    zero_kernel<<<4, 256, 0, stream>>>(stats, 1024);
    stats_c_kernel<<<2048, 256, 0, stream>>>(zA, ROWS_N * 32, stats);
    bn_c_act_kernel<<<cdiv(ROWS_N * 32, 256), 256, 0, stream>>>(
        zA, stats, bn1g, bn1b, zA16, ROWS_N * 32);

    gemm_f16_kernel<<<gemmGrid(250000, 32), 256, 0, stream>>>(
        zA16, wu2, upd2b, zB, (int)ROWS_N, 32, 32);
    zero_kernel<<<4, 256, 0, stream>>>(stats, 1024);
    stats_c_kernel<<<2048, 256, 0, stream>>>(zB, ROWS_N * 32, stats);
    bn_c_resid_kernel<<<cdiv(ROWS_N * 32, 256), 256, 0, stream>>>(
        zB, stats, bn2g, bn2b, h32, h16, ROWS_N * 32);
  }

  // head: h16 viewed as [50000, 160]
  gemm_f16_kernel<<<gemmGrid(50000, 256), 256, 0, stream>>>(
      h16, Wh + 21504, poolB, zA, 50000, 160, 256);
  zero_kernel<<<4, 256, 0, stream>>>(stats, 1024);
  stats_f_kernel<<<2048, 256, 0, stream>>>(zA, 50000L * 256, 256, stats);
  bn_f_act_kernel<<<cdiv(50000L * 256, 256), 256, 0, stream>>>(
      zA, stats, poolBnG, poolBnB, zA16, 50000L * 256, 256, 50000.f);

  gemm_f16_kernel<<<gemmGrid(50000, 128), 256, 0, stream>>>(
      zA16, Wh + 62464, mlp0B, zB, 50000, 256, 128);
  zero_kernel<<<4, 256, 0, stream>>>(stats, 1024);
  stats_f_kernel<<<2048, 256, 0, stream>>>(zB, 50000L * 128, 128, stats);
  bn_f_act_kernel<<<cdiv(50000L * 128, 256), 256, 0, stream>>>(
      zB, stats, mlp0BnG, mlp0BnB, zB16, 50000L * 128, 128, 50000.f);

  gemm_f16_kernel<<<gemmGrid(50000, 64), 256, 0, stream>>>(
      zB16, Wh + 95232, mlp1B, zA, 50000, 128, 64);
  zero_kernel<<<4, 256, 0, stream>>>(stats, 1024);
  stats_f_kernel<<<2048, 256, 0, stream>>>(zA, 50000L * 64, 64, stats);
  bn_f_act_kernel<<<cdiv(50000L * 64, 256), 256, 0, stream>>>(
      zA, stats, mlp1BnG, mlp1BnB, (_Float16*)nullptr, 50000L * 64, 64,
      50000.f);

  mean64_kernel<<<cdiv(50000, 256), 256, 0, stream>>>(zA, (float*)d_out, 50000);
}
